// GIN_57440892617235
// MI455X (gfx1250) — matrix operand order repaired and checked by reading, not compiled, not run on hardware
//
#include <hip/hip_runtime.h>

#define DFEAT 128
#define PITCH 136  // ushort pitch per row: 68 dwords, rows 16B-aligned for b128 LDS loads

typedef __attribute__((ext_vector_type(16))) __bf16    v16bf;
typedef __attribute__((ext_vector_type(8)))  float     v8f;
typedef __attribute__((ext_vector_type(8)))  unsigned  v8u;

// ---- bf16 helpers (manual RNE, bit-level) ----
__device__ __forceinline__ unsigned short f2bf_bits(float f) {
    unsigned u = __builtin_bit_cast(unsigned, f);
    return (unsigned short)((u + 0x7FFFu + ((u >> 16) & 1u)) >> 16);
}
__device__ __forceinline__ float bfbits2f(unsigned short s) {
    unsigned u = ((unsigned)s) << 16;
    return __builtin_bit_cast(float, u);
}
__device__ __forceinline__ __bf16 f2bf(float f) {
    return __builtin_bit_cast(__bf16, f2bf_bits(f));
}

// Pack fp32 pair -> (hiPair dword, loPair dword)
__device__ __forceinline__ void cvt_pair(float x0, float x1, unsigned& hiP, unsigned& loP) {
    unsigned short h0 = f2bf_bits(x0), h1 = f2bf_bits(x1);
    unsigned short l0 = f2bf_bits(x0 - bfbits2f(h0));
    unsigned short l1 = f2bf_bits(x1 - bfbits2f(h1));
    hiP = (unsigned)h0 | ((unsigned)h1 << 16);
    loP = (unsigned)l0 | ((unsigned)l1 << 16);
}

// Load one 16x16x32 bf16 A-fragment from a packed-pair LDS row.
// Lane: M = lane&15 (row), half = lane>>4. Element pair v needs
//   K = kt*32 + 16*(v>=4) + 8*half + 2*(v&3)  ->  pair p = kt*16 + 8*(v>=4) + 4*half + (v&3)
// i.e. two groups of 4 consecutive dwords -> two ds_load_b128.
__device__ __forceinline__ v16bf load_afrag(const unsigned short* __restrict__ row,
                                            int kt, int half) {
    const uint4 q0 = *(const uint4*)(row + (kt * 16 + 4 * half) * 2);
    const uint4 q1 = *(const uint4*)(row + (kt * 16 + 8 + 4 * half) * 2);
    v8u u = {q0.x, q0.y, q0.z, q0.w, q1.x, q1.y, q1.z, q1.w};
    return __builtin_bit_cast(v16bf, u);
}

// ---------------------------------------------------------------------------
// Kernel 1: copy (agg := x), vectorized
// ---------------------------------------------------------------------------
__global__ __launch_bounds__(256) void gin_copy_f4(const float4* __restrict__ src,
                                                   float4* __restrict__ dst, int n4) {
    int i = blockIdx.x * 256 + threadIdx.x;
    int stride = gridDim.x * 256;
    for (; i < n4; i += stride) dst[i] = src[i];
}

// ---------------------------------------------------------------------------
// Kernel 2: edge scatter-add: agg[dst] += x[src]. One wave32 per edge, each
// lane moves a float4 (32*4 = 128 feats). Atomics resolve in L2 (51 MB agg).
// ---------------------------------------------------------------------------
__global__ __launch_bounds__(256) void gin_scatter_add(const float* __restrict__ x,
                                                       const int* __restrict__ srcIdx,
                                                       const int* __restrict__ dstIdx,
                                                       float* __restrict__ agg, int nE) {
    long long t = (long long)blockIdx.x * 256 + threadIdx.x;
    int e = (int)(t >> 5);
    if (e >= nE) return;
    int lane = (int)(t & 31);
    int s = srcIdx[e];
    int d = dstIdx[e];
    const float4 v = *(const float4*)(x + (size_t)s * DFEAT + lane * 4);
    float* p = agg + (size_t)d * DFEAT + lane * 4;
    atomicAdd(p + 0, v.x);
    atomicAdd(p + 1, v.y);
    atomicAdd(p + 2, v.z);
    atomicAdd(p + 3, v.w);
}

// ---------------------------------------------------------------------------
// Kernel 3: fused 2-stage MLP:  Out = act2( act1(A@W1) @ W2 + b2 )
//   stage1 epilogue: h*s1scale[col] + s1shift[col] (bias+BN folded), ReLU
//   stage2 epilogue: + b2[col], optional ReLU
// 256 threads = 8 wave32s; wave w owns output cols [16w,16w+16).
// Tiles live in LDS as PRE-CONVERTED packed bf16 hi/lo pairs, so fp32->bf16
// split conversion happens exactly once per element; the WMMA inner loop is
// pure ds_load_b128 -> v_wmma_f32_16x16x32_bf16 (3-product split-bf16 EFT,
// ~2^-16 relative error vs fp32 reference).
// ---------------------------------------------------------------------------
__global__ __launch_bounds__(256) void gin_mlp2(const float* __restrict__ Ain, int nRows,
                                                const float* __restrict__ W1,
                                                const float* __restrict__ b1,
                                                const float* __restrict__ gamma,
                                                const float* __restrict__ beta,
                                                const float* __restrict__ mean,
                                                const float* __restrict__ var,
                                                const float* __restrict__ W2,
                                                const float* __restrict__ b2,
                                                float* __restrict__ Out, int relu2) {
    __shared__ unsigned short aHi[16 * PITCH], aLo[16 * PITCH];  // input tile
    __shared__ unsigned short tHi[16 * PITCH], tLo[16 * PITCH];  // stage-1 result

    const int tid  = threadIdx.x;
    const int wave = tid >> 5;
    const int lane = tid & 31;
    const int half = lane >> 4;
    const int nrow = lane & 15;
    const int col  = wave * 16 + nrow;  // this lane's output feature (B/C/D: N = lane&15)

    // Per-column epilogue constants (bias + eval BatchNorm folded)
    float s1scale = 1.f;
    float s1shift = b1[col];
    if (gamma) {
        float a = gamma[col] * rsqrtf(var[col] + 1e-5f);
        s1scale = a;
        s1shift = (b1[col] - mean[col]) * a + beta[col];
    }
    const float b2c = b2[col];

    // Preload W1/W2 fragments (hi+lo) once per block; reused over all row tiles.
    // B layout (16x16x32 bf16): lane -> N = lane&15, half = lane>>4,
    // element e -> K = 8*half + e (e<8), 16 + 8*half + (e-8) (e>=8).
    v16bf w1hi[4], w1lo[4], w2hi[4], w2lo[4];
#pragma unroll
    for (int kt = 0; kt < 4; ++kt) {
#pragma unroll
        for (int e = 0; e < 16; ++e) {
            const int k = kt * 32 + (e < 8 ? half * 8 + e : 16 + half * 8 + (e - 8));
            float f1 = W1[k * DFEAT + col];
            __bf16 h1 = f2bf(f1);
            w1hi[kt][e] = h1;
            w1lo[kt][e] = f2bf(f1 - bfbits2f(f2bf_bits(f1)));
            float f2 = W2[k * DFEAT + col];
            __bf16 h2 = f2bf(f2);
            w2hi[kt][e] = h2;
            w2lo[kt][e] = f2bf(f2 - bfbits2f(f2bf_bits(f2)));
        }
    }

    const int nTiles = (nRows + 15) >> 4;
    for (int rt = blockIdx.x; rt < nTiles; rt += gridDim.x) {
        const int rowBase = rt * 16;
        __syncthreads();  // previous iteration's tile reads complete before reuse
        // Stage input tile: load fp32 pairs (coalesced), convert ONCE, store packed
        for (int idx = tid; idx < 16 * 64; idx += 256) {
            const int r = idx >> 6, p = idx & 63;
            const int gr = rowBase + r;
            float2 xv = make_float2(0.f, 0.f);
            if (gr < nRows) xv = *(const float2*)(Ain + (size_t)gr * DFEAT + p * 2);
            unsigned hiP, loP;
            cvt_pair(xv.x, xv.y, hiP, loP);
            *(unsigned*)(aHi + r * PITCH + p * 2) = hiP;
            *(unsigned*)(aLo + r * PITCH + p * 2) = loP;
        }
        __syncthreads();

        // ---- stage 1: acc = A @ W1 (split-bf16, fp32 accum) ----
        const unsigned short* aHiRow = aHi + nrow * PITCH;
        const unsigned short* aLoRow = aLo + nrow * PITCH;
        v8f acc = {};
#pragma unroll
        for (int kt = 0; kt < 4; ++kt) {
            const v16bf ahi = load_afrag(aHiRow, kt, half);
            const v16bf alo = load_afrag(aLoRow, kt, half);
            acc = __builtin_amdgcn_wmma_f32_16x16x32_bf16(false, ahi, false, w1hi[kt],
                                                          (short)0, acc, false, false);
            acc = __builtin_amdgcn_wmma_f32_16x16x32_bf16(false, ahi, false, w1lo[kt],
                                                          (short)0, acc, false, false);
            acc = __builtin_amdgcn_wmma_f32_16x16x32_bf16(false, alo, false, w1hi[kt],
                                                          (short)0, acc, false, false);
        }
        // epilogue 1: fold bias+BN, ReLU, split to bf16 hi/lo -> LDS t tile
        // (C/D layout: element i lives at M = i + 8*half, N = col)
#pragma unroll
        for (int i = 0; i < 8; ++i) {
            float hv = fmaxf(acc[i] * s1scale + s1shift, 0.f);
            unsigned short hb = f2bf_bits(hv);
            unsigned short lb = f2bf_bits(hv - bfbits2f(hb));
            const int m = i + half * 8;
            tHi[m * PITCH + col] = hb;
            tLo[m * PITCH + col] = lb;
        }
        __syncthreads();

        // ---- stage 2: acc2 = t @ W2 ----
        const unsigned short* tHiRow = tHi + nrow * PITCH;
        const unsigned short* tLoRow = tLo + nrow * PITCH;
        v8f acc2 = {};
#pragma unroll
        for (int kt = 0; kt < 4; ++kt) {
            const v16bf ahi = load_afrag(tHiRow, kt, half);
            const v16bf alo = load_afrag(tLoRow, kt, half);
            acc2 = __builtin_amdgcn_wmma_f32_16x16x32_bf16(false, ahi, false, w2hi[kt],
                                                           (short)0, acc2, false, false);
            acc2 = __builtin_amdgcn_wmma_f32_16x16x32_bf16(false, ahi, false, w2lo[kt],
                                                           (short)0, acc2, false, false);
            acc2 = __builtin_amdgcn_wmma_f32_16x16x32_bf16(false, alo, false, w2hi[kt],
                                                           (short)0, acc2, false, false);
        }
        // epilogue 2: bias (+ optional ReLU) -> global
#pragma unroll
        for (int i = 0; i < 8; ++i) {
            float ov = acc2[i] + b2c;
            if (relu2) ov = fmaxf(ov, 0.f);
            const int gr = rowBase + i + half * 8;
            if (gr < nRows) Out[(size_t)gr * DFEAT + col] = ov;
        }
    }
}

// ---------------------------------------------------------------------------
// Kernel 4: zero a float buffer
// ---------------------------------------------------------------------------
__global__ __launch_bounds__(256) void gin_zero(float* __restrict__ p, int n) {
    int i = blockIdx.x * 256 + threadIdx.x;
    if (i < n) p[i] = 0.f;
}

// ---------------------------------------------------------------------------
// Kernel 5: global_add_pool: g[batch[i]] += x[i]. Wave per node.
// ---------------------------------------------------------------------------
__global__ __launch_bounds__(256) void gin_pool(const float* __restrict__ x,
                                                const int* __restrict__ batch,
                                                float* __restrict__ g, int nNodes) {
    long long t = (long long)blockIdx.x * 256 + threadIdx.x;
    int node = (int)(t >> 5);
    if (node >= nNodes) return;
    int lane = (int)(t & 31);
    int b = batch[node];
    const float4 v = *(const float4*)(x + (size_t)node * DFEAT + lane * 4);
    float* p = g + (size_t)b * DFEAT + lane * 4;
    atomicAdd(p + 0, v.x);
    atomicAdd(p + 1, v.y);
    atomicAdd(p + 2, v.z);
    atomicAdd(p + 3, v.w);
}

// ---------------------------------------------------------------------------
extern "C" void kernel_launch(void* const* d_in, const int* in_sizes, int n_in,
                              void* d_out, int out_size, void* d_ws, size_t ws_size,
                              hipStream_t stream) {
    const float* x     = (const float*)d_in[0];
    const int*   eidx  = (const int*)d_in[1];
    const int*   batch = (const int*)d_in[2];
    const float* cW1   = (const float*)d_in[3];
    const float* cb1   = (const float*)d_in[4];
    const float* bnG   = (const float*)d_in[5];
    const float* bnB   = (const float*)d_in[6];
    const float* bnM   = (const float*)d_in[7];
    const float* bnV   = (const float*)d_in[8];
    const float* cW2   = (const float*)d_in[9];
    const float* cb2   = (const float*)d_in[10];
    const float* hW1   = (const float*)d_in[11];
    const float* hb1   = (const float*)d_in[12];
    const float* hW2   = (const float*)d_in[13];
    const float* hb2   = (const float*)d_in[14];
    float*       out   = (float*)d_out;

    const int N = in_sizes[0] / DFEAT;  // nodes
    const int E = in_sizes[1] / 2;      // edges
    const int G = out_size / DFEAT;     // graphs
    const int L = in_sizes[3] / (DFEAT * DFEAT);

    const int* src = eidx;
    const int* dst = eidx + E;

    const size_t ndBytes = (size_t)N * DFEAT * sizeof(float);
    char* ws = (char*)d_ws;
    float* agg = (float*)(ws);
    float* xb0 = (float*)(ws + ndBytes);
    float* xb1 = (float*)(ws + 2 * ndBytes);
    float* g   = (float*)(ws + 3 * ndBytes);

    const int n4 = (N * DFEAT) / 4;
    const int copyBlocks = 2048;
    const int scatBlocks = (int)(((long long)E * 32 + 255) / 256);
    const int poolBlocks = (int)(((long long)N * 32 + 255) / 256);
    const int nodeTiles  = (N + 15) / 16;
    const int gemmBlocks = nodeTiles < 2048 ? nodeTiles : 2048;
    const int headTiles  = (G + 15) / 16;

    const float* xcur = x;
    float* ping[2] = {xb0, xb1};
    for (int l = 0; l < L; ++l) {
        // agg = x + sum_{j->i} x_j
        gin_copy_f4<<<copyBlocks, 256, 0, stream>>>((const float4*)xcur, (float4*)agg, n4);
        gin_scatter_add<<<scatBlocks, 256, 0, stream>>>(xcur, src, dst, agg, E);
        // x = relu( relu(BN(agg@W1 + b1)) @ W2 + b2 )
        float* xnext = ping[l & 1];
        gin_mlp2<<<gemmBlocks, 256, 0, stream>>>(
            agg, N, cW1 + (size_t)l * DFEAT * DFEAT, cb1 + (size_t)l * DFEAT,
            bnG + (size_t)l * DFEAT, bnB + (size_t)l * DFEAT,
            bnM + (size_t)l * DFEAT, bnV + (size_t)l * DFEAT,
            cW2 + (size_t)l * DFEAT * DFEAT, cb2 + (size_t)l * DFEAT, xnext, 1);
        xcur = xnext;
    }

    // global_add_pool
    gin_zero<<<(G * DFEAT + 255) / 256, 256, 0, stream>>>(g, G * DFEAT);
    gin_pool<<<poolBlocks, 256, 0, stream>>>(xcur, batch, g, N);

    // head: relu(g@hW1 + hb1) @ hW2 + hb2  (no BN, no final relu)
    gin_mlp2<<<headTiles, 256, 0, stream>>>(g, G, hW1, hb1, nullptr, nullptr, nullptr,
                                            nullptr, hW2, hb2, out, 0);
}
